// SelfAttention_34179349741637
// MI455X (gfx1250) — matrix operand order repaired and checked by reading, not compiled, not run on hardware
//
#include <hip/hip_runtime.h>
#include <hip/hip_bf16.h>

// Fixed problem size from the reference
#define BB 8
#define SS 2048
#define HH 512

typedef __attribute__((ext_vector_type(16))) __bf16 v16bf;
typedef __attribute__((ext_vector_type(8)))  float  v8f;

union FragU {
  v16bf v;
  uint4 q[2];
};

__device__ __forceinline__ v8f wmma_bf16(v16bf a, v16bf b, v8f c) {
  // v_wmma_f32_16x16x32_bf16: (neg_a, A, neg_b, B, c_mod, C, reuse_a, reuse_b)
  return __builtin_amdgcn_wmma_f32_16x16x32_bf16(false, a, false, b, (short)0, c,
                                                 false, false);
}

// ---- Fragment loaders ------------------------------------------------------
// A-matrix 16x32 bf16 layout (per ISA table):
//   lanes 0-15 (half=0):  v[0..7]=K 0..7,  v[8..15]=K 16..23  (row M = lane)
//   lanes 16-31 (half=1): v[0..7]=K 8..15, v[8..15]=K 24..31  (row M = lane-16)
__device__ __forceinline__ v16bf load_a_bf16(const __bf16* __restrict__ row,
                                             int k0, int half) {
  FragU f;
  f.q[0] = *(const uint4*)(row + k0 + half * 8);
  f.q[1] = *(const uint4*)(row + k0 + 16 + half * 8);
  return f.v;
}

__device__ __forceinline__ v16bf load_a_f32(const float* __restrict__ row,
                                            int k0, int half) {
  FragU f;
  const float* p0 = row + k0 + half * 8;
  const float* p1 = row + k0 + 16 + half * 8;
#pragma unroll
  for (int j = 0; j < 8; ++j) {
    f.v[j]     = (__bf16)p0[j];   // native v_cvt_pk_bf16_f32 path
    f.v[8 + j] = (__bf16)p1[j];
  }
  return f.v;
}

// B-matrix 32x16 bf16 layout (col n = lane&15 per lane):
//   lanes 0-15:  v[0..7] = K 0..7,  v[8..15] = K 16..23
//   lanes 16-31: v[0..7] = K 8..15, v[8..15] = K 24..31
__device__ __forceinline__ v16bf load_b_bf16(const __bf16* __restrict__ row,
                                             int k0, int half) {
  FragU f;
  f.q[0] = *(const uint4*)(row + k0 + half * 8);
  f.q[1] = *(const uint4*)(row + k0 + 16 + half * 8);
  return f.v;
}

// ---- Kernel 0: one-time fp32 -> bf16 conversion of x and Wq/Wk/Wv ---------
// 8 elements per thread: 2x float4 load, 4x v_cvt_pk_bf16_f32, 1x b128 store.
__global__ __launch_bounds__(256) void attn_cvt_kernel(
    const float* __restrict__ x, const float* __restrict__ Wq,
    const float* __restrict__ Wk, const float* __restrict__ Wv,
    __bf16* __restrict__ xb, __bf16* __restrict__ Wb) {
  const size_t XN = (size_t)BB * SS * HH;  // 8388608
  const size_t WN = (size_t)HH * HH;       // 262144
  size_t i = ((size_t)blockIdx.x * 256 + threadIdx.x) * 8;

  const float* src;
  __bf16* dst;
  size_t off;
  if (i < XN) {
    src = x; dst = xb; off = i;
  } else {
    size_t r = i - XN;
    int w = (int)(r / WN);  // 0..2
    off = r % WN;
    src = (w == 0) ? Wq : (w == 1) ? Wk : Wv;
    dst = Wb + (size_t)w * WN;
  }

  float4 f0 = *(const float4*)(src + off);
  float4 f1 = *(const float4*)(src + off + 4);
  union { __bf16 b[8]; uint4 u; } o;
  o.b[0] = (__bf16)f0.x; o.b[1] = (__bf16)f0.y;
  o.b[2] = (__bf16)f0.z; o.b[3] = (__bf16)f0.w;
  o.b[4] = (__bf16)f1.x; o.b[5] = (__bf16)f1.y;
  o.b[6] = (__bf16)f1.z; o.b[7] = (__bf16)f1.w;
  *(uint4*)(dst + off) = o.u;
}

// ---- Kernel 1: QKV projection, y = xb @ Wb^T (NT GEMM), 32x32 tile/wave ---
// matId 0 -> Qb (bf16, row-major [B*S, H])
// matId 1 -> Kb (bf16, row-major [B*S, H])
// matId 2 -> Vt (bf16, transposed [B][H][S])
__global__ __launch_bounds__(256) void attn_qkv_kernel(
    const __bf16* __restrict__ xb, const __bf16* __restrict__ Wb,
    __bf16* __restrict__ Qb, __bf16* __restrict__ Kb,
    __bf16* __restrict__ Vt) {
  const int MT = (BB * SS) / 32;  // 512
  const int NT = HH / 32;         // 16
  int lane = threadIdx.x & 31;
  int gw   = blockIdx.x * 8 + (threadIdx.x >> 5);
  int matId = gw / (MT * NT);
  int rem   = gw % (MT * NT);
  int mt = rem / NT, nt = rem % NT;
  int m = lane & 15, half = lane >> 4;

  const __bf16* W = Wb + (size_t)matId * HH * HH;
  const __bf16* arow0 = xb + (size_t)(mt * 32 + m) * HH;
  const __bf16* arow1 = arow0 + (size_t)16 * HH;
  const __bf16* brow0 = W + (size_t)(nt * 32 + m) * HH;
  const __bf16* brow1 = brow0 + (size_t)16 * HH;

  v8f c00 = {0.f,0.f,0.f,0.f,0.f,0.f,0.f,0.f}, c01 = c00, c10 = c00, c11 = c00;
#pragma unroll 4
  for (int k0 = 0; k0 < HH; k0 += 32) {
    v16bf a0 = load_a_bf16(arow0, k0, half);
    v16bf a1 = load_a_bf16(arow1, k0, half);
    v16bf b0 = load_b_bf16(brow0, k0, half);
    v16bf b1 = load_b_bf16(brow1, k0, half);
    c00 = wmma_bf16(a0, b0, c00);
    c01 = wmma_bf16(a0, b1, c01);
    c10 = wmma_bf16(a1, b0, c10);
    c11 = wmma_bf16(a1, b1, c11);
  }

  int col0 = nt * 32 + m, col1 = col0 + 16;
  if (matId < 2) {
    __bf16* out = (matId == 0) ? Qb : Kb;
#pragma unroll
    for (int j = 0; j < 8; ++j) {
      int g0 = mt * 32 + j + 8 * half;
      int g1 = g0 + 16;
      out[(size_t)g0 * HH + col0] = (__bf16)c00[j];
      out[(size_t)g0 * HH + col1] = (__bf16)c01[j];
      out[(size_t)g1 * HH + col0] = (__bf16)c10[j];
      out[(size_t)g1 * HH + col1] = (__bf16)c11[j];
    }
  } else {
#pragma unroll
    for (int j = 0; j < 8; ++j) {
      int g0 = mt * 32 + j + 8 * half;
      int g1 = g0 + 16;
      int b0_ = g0 >> 11, s0_ = g0 & (SS - 1);
      int b1_ = g1 >> 11, s1_ = g1 & (SS - 1);
      Vt[(size_t)b0_ * HH * SS + (size_t)col0 * SS + s0_] = (__bf16)c00[j];
      Vt[(size_t)b0_ * HH * SS + (size_t)col1 * SS + s0_] = (__bf16)c01[j];
      Vt[(size_t)b1_ * HH * SS + (size_t)col0 * SS + s1_] = (__bf16)c10[j];
      Vt[(size_t)b1_ * HH * SS + (size_t)col1 * SS + s1_] = (__bf16)c11[j];
    }
  }
}

// ---- Kernel 2: scores = (Q @ K^T) * 1/sqrt(H), per batch, 32x32/wave ------
__global__ __launch_bounds__(256) void attn_scores_kernel(
    const __bf16* __restrict__ Qb, const __bf16* __restrict__ Kb,
    float* __restrict__ attnW) {
  const int MT = SS / 32;  // 64
  const int NT = SS / 32;  // 64
  const float scale = 0.044194173824159216f;  // 1/sqrt(512)
  int lane = threadIdx.x & 31;
  int gw   = blockIdx.x * 8 + (threadIdx.x >> 5);
  int b_  = gw / (MT * NT);
  int rem = gw % (MT * NT);
  int mt = rem / NT, nt = rem % NT;
  int m = lane & 15, half = lane >> 4;

  const __bf16* arow0 = Qb + (size_t)(b_ * SS + mt * 32 + m) * HH;
  const __bf16* arow1 = arow0 + (size_t)16 * HH;
  const __bf16* brow0 = Kb + (size_t)(b_ * SS + nt * 32 + m) * HH;
  const __bf16* brow1 = brow0 + (size_t)16 * HH;

  v8f c00 = {0.f,0.f,0.f,0.f,0.f,0.f,0.f,0.f}, c01 = c00, c10 = c00, c11 = c00;
#pragma unroll 4
  for (int k0 = 0; k0 < HH; k0 += 32) {
    v16bf a0 = load_a_bf16(arow0, k0, half);
    v16bf a1 = load_a_bf16(arow1, k0, half);
    v16bf b0 = load_b_bf16(brow0, k0, half);
    v16bf b1 = load_b_bf16(brow1, k0, half);
    c00 = wmma_bf16(a0, b0, c00);
    c01 = wmma_bf16(a0, b1, c01);
    c10 = wmma_bf16(a1, b0, c10);
    c11 = wmma_bf16(a1, b1, c11);
  }

  float* obase = attnW + (size_t)b_ * SS * SS;
  int col0 = nt * 32 + m, col1 = col0 + 16;
#pragma unroll
  for (int j = 0; j < 8; ++j) {
    int q0 = mt * 32 + j + 8 * half;
    int q1 = q0 + 16;
    obase[(size_t)q0 * SS + col0] = c00[j] * scale;
    obase[(size_t)q0 * SS + col1] = c01[j] * scale;
    obase[(size_t)q1 * SS + col0] = c10[j] * scale;
    obase[(size_t)q1 * SS + col1] = c11[j] * scale;
  }
}

// ---- Kernel 3: row-wise softmax in place, one block per row ---------------
__global__ __launch_bounds__(256) void attn_softmax_kernel(float* __restrict__ P) {
  __shared__ float sm[256];
  int t = threadIdx.x;
  float* row = P + (size_t)blockIdx.x * SS;

  float v[8];
#pragma unroll
  for (int i = 0; i < 8; ++i) v[i] = row[t + i * 256];

  float mx = v[0];
#pragma unroll
  for (int i = 1; i < 8; ++i) mx = fmaxf(mx, v[i]);
  sm[t] = mx;
  __syncthreads();
  for (int off = 128; off > 0; off >>= 1) {
    if (t < off) sm[t] = fmaxf(sm[t], sm[t + off]);
    __syncthreads();
  }
  mx = sm[0];
  __syncthreads();

  float s = 0.f;
#pragma unroll
  for (int i = 0; i < 8; ++i) {
    v[i] = __expf(v[i] - mx);
    s += v[i];
  }
  sm[t] = s;
  __syncthreads();
  for (int off = 128; off > 0; off >>= 1) {
    if (t < off) sm[t] += sm[t + off];
    __syncthreads();
  }
  float inv = 1.0f / sm[0];

#pragma unroll
  for (int i = 0; i < 8; ++i) row[t + i * 256] = v[i] * inv;
}

// ---- Kernel 4: attn_output = P @ V  (A = P fp32->bf16, B = Vt), 32x32 -----
__global__ __launch_bounds__(256) void attn_out_kernel(
    const float* __restrict__ P, const __bf16* __restrict__ Vt,
    float* __restrict__ out) {
  const int MT = SS / 32;  // 64
  const int NT = HH / 32;  // 16
  int lane = threadIdx.x & 31;
  int gw   = blockIdx.x * 8 + (threadIdx.x >> 5);
  int b_  = gw / (MT * NT);
  int rem = gw % (MT * NT);
  int mt = rem / NT, nt = rem % NT;
  int m = lane & 15, half = lane >> 4;

  const float* arow0 = P + (size_t)b_ * SS * SS + (size_t)(mt * 32 + m) * SS;
  const float* arow1 = arow0 + (size_t)16 * SS;
  const __bf16* brow0 = Vt + (size_t)b_ * HH * SS + (size_t)(nt * 32 + m) * SS;
  const __bf16* brow1 = brow0 + (size_t)16 * SS;

  v8f c00 = {0.f,0.f,0.f,0.f,0.f,0.f,0.f,0.f}, c01 = c00, c10 = c00, c11 = c00;
#pragma unroll 2
  for (int k0 = 0; k0 < SS; k0 += 32) {
    v16bf a0 = load_a_f32(arow0, k0, half);
    v16bf a1 = load_a_f32(arow1, k0, half);
    v16bf b0 = load_b_bf16(brow0, k0, half);
    v16bf b1 = load_b_bf16(brow1, k0, half);
    c00 = wmma_bf16(a0, b0, c00);
    c01 = wmma_bf16(a0, b1, c01);
    c10 = wmma_bf16(a1, b0, c10);
    c11 = wmma_bf16(a1, b1, c11);
  }

  int col0 = nt * 32 + m, col1 = col0 + 16;
#pragma unroll
  for (int j = 0; j < 8; ++j) {
    int q0 = mt * 32 + j + 8 * half;
    int q1 = q0 + 16;
    out[(size_t)(b_ * SS + q0) * HH + col0] = c00[j];
    out[(size_t)(b_ * SS + q0) * HH + col1] = c01[j];
    out[(size_t)(b_ * SS + q1) * HH + col0] = c10[j];
    out[(size_t)(b_ * SS + q1) * HH + col1] = c11[j];
  }
}

extern "C" void kernel_launch(void* const* d_in, const int* in_sizes, int n_in,
                              void* d_out, int out_size, void* d_ws, size_t ws_size,
                              hipStream_t stream) {
  const float* x  = (const float*)d_in[0];
  const float* Wq = (const float*)d_in[1];
  const float* Wk = (const float*)d_in[2];
  const float* Wv = (const float*)d_in[3];

  float* attn_out = (float*)d_out;                    // [B,S,H]
  float* attn_w   = attn_out + (size_t)BB * SS * HH;  // [B,S,S]

  __bf16* Qb = (__bf16*)d_ws;                         // [B*S, H]
  __bf16* Kb = Qb + (size_t)BB * SS * HH;             // [B*S, H]
  __bf16* Vt = Kb + (size_t)BB * SS * HH;             // [B, H, S]
  __bf16* xb = Vt + (size_t)BB * SS * HH;             // [B*S, H]
  __bf16* Wb = xb + (size_t)BB * SS * HH;             // [3, H, H]

  const size_t CVT_ELEMS = (size_t)BB * SS * HH + 3 * (size_t)HH * HH;
  const int CVT_BLOCKS = (int)(CVT_ELEMS / (8 * 256));              // 4480
  const int QKV_BLOCKS = 3 * ((BB * SS) / 32) * (HH / 32) / 8;      // 3072
  const int SC_BLOCKS  = BB * (SS / 32) * (SS / 32) / 8;            // 4096
  const int SM_BLOCKS  = BB * SS;                                   // 16384
  const int OUT_BLOCKS = BB * (SS / 32) * (HH / 32) / 8;            // 1024

  attn_cvt_kernel<<<CVT_BLOCKS, 256, 0, stream>>>(x, Wq, Wk, Wv, xb, Wb);
  attn_qkv_kernel<<<QKV_BLOCKS, 256, 0, stream>>>(xb, Wb, Qb, Kb, Vt);
  attn_scores_kernel<<<SC_BLOCKS, 256, 0, stream>>>(Qb, Kb, attn_w);
  attn_softmax_kernel<<<SM_BLOCKS, 256, 0, stream>>>(attn_w);
  attn_out_kernel<<<OUT_BLOCKS, 256, 0, stream>>>(attn_w, Vt, attn_out);
}
